// GCN_678604832909
// MI455X (gfx1250) — matrix operand order repaired and checked by reading, not compiled, not run on hardware
//
#include <hip/hip_runtime.h>
#include <hip/hip_bf16.h>

typedef __attribute__((ext_vector_type(16))) __bf16 v16bf;
typedef __attribute__((ext_vector_type(8)))  __bf16 v8bf;
typedef __attribute__((ext_vector_type(8)))  float  v8f;

#define NFEAT 512
#define NHID  256
#define NCLASS 40

// ---------------------------------------------------------------------------
// Zero a float buffer (float4-wide).
__global__ __launch_bounds__(256) void zero_f4(float4* __restrict__ p, size_t n4) {
  size_t i = (size_t)blockIdx.x * blockDim.x + threadIdx.x;
  size_t stride = (size_t)gridDim.x * blockDim.x;
  for (; i < n4; i += stride) p[i] = make_float4(0.f, 0.f, 0.f, 0.f);
}

// ---------------------------------------------------------------------------
// C[M x NSTORE] (bf16) = A[M x K] (TIn) @ W[K x NSTORE] (f32), f32 accumulate.
// One wave handles a 16-row tile across ALL column tiles so A is read once.
// W is staged in LDS as bf16 in a k-chunked layout:
//   Wl[ ((k>>4) * LDSN + n) * 16 + (k&15) ]
// so a B fragment (16 contiguous k for one n) is one 32B read, the nt-stride
// is 512B (immediate DS offsets across all column tiles), ks-stride is 16KB.
template <typename TIn, int K, int NT, int NSTORE>
__global__ __launch_bounds__(256) void gemm_wmma(const TIn* __restrict__ A,
                                                 const float* __restrict__ W,
                                                 __bf16* __restrict__ C, int M) {
  constexpr int LDSN = NT * 16;
  constexpr int KS = K / 32;
  extern __shared__ char smem_raw[];
  __bf16* Wl = (__bf16*)smem_raw;

  const int tid = threadIdx.x;
  for (int idx = tid; idx < K * NSTORE; idx += 256) {
    int k = idx / NSTORE;
    int n = idx - k * NSTORE;
    Wl[(((k >> 4) * LDSN + n) << 4) + (k & 15)] = (__bf16)W[idx];
  }
  if constexpr (LDSN > NSTORE) {  // zero-pad the tail column tile
    for (int idx = tid; idx < K * (LDSN - NSTORE); idx += 256) {
      int k = idx / (LDSN - NSTORE);
      int n = NSTORE + (idx - k * (LDSN - NSTORE));
      Wl[(((k >> 4) * LDSN + n) << 4) + (k & 15)] = (__bf16)0.0f;
    }
  }
  __syncthreads();

  const int lane  = tid & 31;
  const int wave  = tid >> 5;
  const int l16   = lane & 15;   // M-row within tile (A), N-col within tile (B/C)
  const int khalf = lane >> 4;   // which K half this lane holds
  const int Mtiles = (M + 15) >> 4;
  const int nwaves = gridDim.x * 8;

  for (int mt = blockIdx.x * 8 + wave; mt < Mtiles; mt += nwaves) {
    int row = mt * 16 + l16;
    if (row >= M) row = M - 1;  // clamp for tail tile; store is guarded below
    const TIn* arow = A + (size_t)row * K;

    // A fragment: lane half 0 holds K ks*32..+15, half 1 holds +16..+31
    // (contiguous, matching the B fragment).
    auto loadA = [&](int ks) -> v16bf {
      const int k0 = ks * 32 + khalf * 16;
      v16bf af;
#pragma unroll
      for (int e = 0; e < 8; ++e) {
        af[e]     = (__bf16)(float)arow[k0 + e];
        af[e + 8] = (__bf16)(float)arow[k0 + 8 + e];
      }
      return af;
    };
    // B fragment: lanes 0-15 hold K ks*32..+15, lanes 16-31 hold +16..+31.
    auto loadB = [&](int ks, int nt) -> v16bf {
      const int c = ks * 2 + khalf;
      return *(const v16bf*)&Wl[(c * LDSN + nt * 16 + l16) << 4];
    };

    v8f acc[NT] = {};

    // Software pipeline: A one k-step ahead, B two fragments ahead in a
    // rotating 3-buffer ring (avoids exposed s_wait_dscnt 0 and the
    // WMMA->ds_load WAR-hazard NOPs from register reuse).
    v16bf af = loadA(0);
    v16bf b0 = loadB(0, 0);
    v16bf b1 = (NT > 1) ? loadB(0, 1) : loadB(0, 0);

#pragma unroll 2
    for (int ks = 0; ks < KS; ++ks) {
      v16bf af_n = af;
      if (ks + 1 < KS) af_n = loadA(ks + 1);
#pragma unroll
      for (int nt = 0; nt < NT; ++nt) {
        int pn = nt + 2, pk = ks;           // fragment two WMMAs ahead
        if (pn >= NT) { pn -= NT; ++pk; }
        if (pn >= NT) { pn -= NT; ++pk; }   // (covers NT==1; folds away)
        v16bf b2 = b0;
        if (pk < KS) b2 = loadB(pk, pn);
        acc[nt] = __builtin_amdgcn_wmma_f32_16x16x32_bf16(
            false, af, false, b0, (short)0, acc[nt], false, false);
        b0 = b1;
        b1 = b2;
      }
      af = af_n;
    }

    // C/D layout: VGPR r = row (khalf*8 + r), lane halves split M 0-7 / 8-15.
    const int rbase = mt * 16 + khalf * 8;
#pragma unroll
    for (int nt = 0; nt < NT; ++nt) {
      const int n = nt * 16 + l16;
      if (LDSN != NSTORE && n >= NSTORE) continue;
#pragma unroll
      for (int r = 0; r < 8; ++r) {
        const int rr = rbase + r;
        if (rr < M) C[(size_t)rr * NSTORE + n] = (__bf16)acc[nt][r];
      }
    }
  }
}

// ---------------------------------------------------------------------------
// h = bf16(relu(agg + b)), D = NHID (power of two).
__global__ __launch_bounds__(256) void bias_relu_to_bf16(
    const float* __restrict__ agg, const float* __restrict__ b,
    __bf16* __restrict__ h, size_t total) {
  size_t i = (size_t)blockIdx.x * blockDim.x + threadIdx.x;
  size_t stride = (size_t)gridDim.x * blockDim.x;
  for (; i < total; i += stride) {
    float v = agg[i] + b[i & (NHID - 1)];
    h[i] = (__bf16)fmaxf(v, 0.0f);
  }
}

// ---------------------------------------------------------------------------
// One wave per edge: agg[dst] += support[src] * w.  D = 256: 8 bf16 per lane
// (one 16B load) -> 8 f32 atomic adds.  agg (102 MB) lives in L2.
__global__ __launch_bounds__(256) void spmm_edges_d256(
    const __bf16* __restrict__ S, const int* __restrict__ src,
    const int* __restrict__ dst, const float* __restrict__ w,
    float* __restrict__ agg, int E) {
  const int lane = threadIdx.x & 31;
  const int e = (int)((blockIdx.x * 256u + threadIdx.x) >> 5);
  if (e >= E) return;
  const int s = src[e], d = dst[e];
  const float ww = w[e];
  const v8bf v = *(const v8bf*)(S + (size_t)s * NHID + lane * 8);
  float* arow = agg + (size_t)d * NHID + lane * 8;
#pragma unroll
  for (int j = 0; j < 8; ++j) atomicAdd(arow + j, (float)v[j] * ww);
}

// One wave per edge, D = 40: lanes 0..19 handle 2 elements each.
__global__ __launch_bounds__(256) void spmm_edges_d40(
    const __bf16* __restrict__ S, const int* __restrict__ src,
    const int* __restrict__ dst, const float* __restrict__ w,
    float* __restrict__ agg, int E) {
  const int lane = threadIdx.x & 31;
  const int e = (int)((blockIdx.x * 256u + threadIdx.x) >> 5);
  if (e >= E) return;
  const int s = src[e], d = dst[e];
  const float ww = w[e];
  if (lane < 20) {
    const __bf16* sp = S + (size_t)s * NCLASS + lane * 2;
    float* ap = agg + (size_t)d * NCLASS + lane * 2;
    atomicAdd(ap,     (float)sp[0] * ww);
    atomicAdd(ap + 1, (float)sp[1] * ww);
  }
}

// ---------------------------------------------------------------------------
// out = log_softmax(agg + b) over 40 classes; one wave per node, wave32
// shuffle reductions.
__global__ __launch_bounds__(256) void bias_logsoftmax40(
    const float* __restrict__ agg, const float* __restrict__ b,
    float* __restrict__ out, int N) {
  const int lane = threadIdx.x & 31;
  const int node = (int)((blockIdx.x * 256u + threadIdx.x) >> 5);
  if (node >= N) return;
  const float* row = agg + (size_t)node * NCLASS;
  const int c0 = lane * 2, c1 = c0 + 1;
  float x0 = -3.0e38f, x1 = -3.0e38f;
  if (lane < 20) {
    x0 = row[c0] + b[c0];
    x1 = row[c1] + b[c1];
  }
  float m = fmaxf(x0, x1);
#pragma unroll
  for (int off = 16; off; off >>= 1) m = fmaxf(m, __shfl_xor(m, off, 32));
  float s = (lane < 20) ? (__expf(x0 - m) + __expf(x1 - m)) : 0.0f;
#pragma unroll
  for (int off = 16; off; off >>= 1) s += __shfl_xor(s, off, 32);
  const float lse = m + __logf(s);
  if (lane < 20) {
    out[(size_t)node * NCLASS + c0] = x0 - lse;
    out[(size_t)node * NCLASS + c1] = x1 - lse;
  }
}

// ---------------------------------------------------------------------------
extern "C" void kernel_launch(void* const* d_in, const int* in_sizes, int n_in,
                              void* d_out, int out_size, void* d_ws, size_t ws_size,
                              hipStream_t stream) {
  const float* x  = (const float*)d_in[0];
  const int*   ei = (const int*)d_in[1];   // [2, E]
  const float* ew = (const float*)d_in[2];
  const float* W1 = (const float*)d_in[3];
  const float* b1 = (const float*)d_in[4];
  const float* W2 = (const float*)d_in[5];
  const float* b2 = (const float*)d_in[6];
  float* out = (float*)d_out;

  const int N = in_sizes[0] / NFEAT;   // 100000
  const int E = in_sizes[2];           // 3200000
  const int* src = ei;
  const int* dst = ei + E;

  // Workspace layout (≈154 MB):
  //   A: support1 / h1  bf16 [N x 256]  (51.2 MB)
  //   B: agg1 f32 [N x 256] (102.4 MB), recycled after ReLU for:
  //        support2 bf16 [N x 40] (8 MB) + agg2 f32 [N x 40] (16 MB)
  char* ws = (char*)d_ws;
  __bf16* support1 = (__bf16*)ws;
  const size_t offB = (size_t)N * NHID * sizeof(__bf16);
  float*  agg1 = (float*)(ws + offB);
  __bf16* support2 = (__bf16*)(ws + offB);
  const size_t s2_bytes = (((size_t)N * NCLASS * sizeof(__bf16)) + 255) & ~(size_t)255;
  float*  agg2 = (float*)(ws + offB + s2_bytes);

  constexpr int SH1 = NHID * NFEAT * 2;      // 256 KB bf16 W1 (chunked) in LDS
  constexpr int SH2 = 48 * NHID * 2;         // 24 KB  bf16 W2 (padded to 48)
  (void)hipFuncSetAttribute(
      reinterpret_cast<const void*>(gemm_wmma<float, NFEAT, NHID / 16, NHID>),
      hipFuncAttributeMaxDynamicSharedMemorySize, SH1);
  (void)hipFuncSetAttribute(
      reinterpret_cast<const void*>(gemm_wmma<__bf16, NHID, 3, NCLASS>),
      hipFuncAttributeMaxDynamicSharedMemorySize, SH2);

  // Layer 1
  gemm_wmma<float, NFEAT, NHID / 16, NHID>
      <<<784, 256, SH1, stream>>>(x, W1, support1, N);
  zero_f4<<<4096, 256, 0, stream>>>((float4*)agg1, (size_t)N * NHID / 4);
  spmm_edges_d256<<<(E + 7) / 8, 256, 0, stream>>>(support1, src, dst, ew, agg1, E);
  bias_relu_to_bf16<<<8192, 256, 0, stream>>>(agg1, b1, support1, (size_t)N * NHID);

  // Layer 2
  gemm_wmma<__bf16, NHID, 3, NCLASS>
      <<<784, 256, SH2, stream>>>(support1, W2, support2, N);
  zero_f4<<<2048, 256, 0, stream>>>((float4*)agg2, (size_t)N * NCLASS / 4);
  spmm_edges_d40<<<(E + 7) / 8, 256, 0, stream>>>(support2, src, dst, ew, agg2, E);
  bias_logsoftmax40<<<(N + 7) / 8, 256, 0, stream>>>(agg2, b2, out, N);
}
